// CareerTreeModel_30889404793607
// MI455X (gfx1250) — matrix operand order repaired and checked by reading, not compiled, not run on hardware
//
#include <hip/hip_runtime.h>
#include <hip/hip_bf16.h>

// ---------------------------------------------------------------------------
// CDNA5 (gfx1250) GraphSAGE + edge MLP.
// bf16 WMMA (v_wmma_f32_16x16x32_bf16) with fp32 accumulation; fp32 atomic
// scatter for the mean aggregation; gathered operands kept in bf16 to halve
// the dominant HBM gather traffic (x[src], h[src], z[pairs]).
// ---------------------------------------------------------------------------

typedef __attribute__((ext_vector_type(16))) __bf16 v16bf;
typedef __attribute__((ext_vector_type(8)))  float  v8f;

__device__ __forceinline__ v8f wmma_bf16(v16bf a, v16bf b, v8f c) {
  return __builtin_amdgcn_wmma_f32_16x16x32_bf16(
      /*neg_a=*/false, a, /*neg_b=*/false, b,
      /*c_mod=*/(short)0, c, /*reuse_a=*/false, /*reuse_b=*/false);
}

// A fragment (16x32) from an f32 row (used for the atomic-accumulated mean
// operand), scaled by 1/degree while converting to bf16.
// ISA layout: lanes 0-15: M=lane, K={k0..k0+7, k0+16..k0+23};
//             lanes 16-31: M=lane-16, K={k0+8..k0+15, k0+24..k0+31}.
__device__ __forceinline__ v16bf a_frag_f32(const float* rowp, float scale, int lane) {
  const float* p = rowp + ((lane & 16) ? 8 : 0);
  v16bf a;
#pragma unroll
  for (int e = 0; e < 8; ++e) a[e] = (__bf16)(p[e] * scale);
#pragma unroll
  for (int e = 0; e < 8; ++e) a[e + 8] = (__bf16)(p[e + 16] * scale);
  return a;
}

// A fragment (16x32) directly from a bf16 row: two contiguous 16B chunks.
__device__ __forceinline__ v16bf a_frag_bf(const __bf16* rowp, int lane) {
  const __bf16* p = rowp + ((lane & 16) ? 8 : 0);
  v16bf a;
#pragma unroll
  for (int e = 0; e < 8; ++e) a[e] = p[e];
#pragma unroll
  for (int e = 0; e < 8; ++e) a[e + 8] = p[e + 16];
  return a;
}

// B fragment (32x16) transposed from a row-major bf16 weight W[outdim][K]:
// B[k][n] = W[col0+n][k].  Two contiguous 16B chunks per lane (same K order as A).
__device__ __forceinline__ v16bf b_frag_bf(const __bf16* __restrict__ W, int ldw,
                                           int col0, int k0, int lane) {
  const __bf16* p = W + (size_t)(col0 + (lane & 15)) * ldw + k0 + ((lane & 16) ? 8 : 0);
  v16bf b;
#pragma unroll
  for (int e = 0; e < 8; ++e) b[e] = p[e];
#pragma unroll
  for (int e = 0; e < 8; ++e) b[e + 8] = p[e + 16];
  return b;
}

// ---------------------------------------------------------------------------
// Prep kernels.
// ---------------------------------------------------------------------------
__global__ void zero_f32(float* __restrict__ p, long n) {
  long i = (long)blockIdx.x * blockDim.x + threadIdx.x;
  long stride = (long)gridDim.x * blockDim.x;
  for (; i < n; i += stride) p[i] = 0.0f;
}

__global__ void cvt_f32_to_bf16(const float* __restrict__ in, __bf16* __restrict__ out,
                                long n) {
  long i = (long)blockIdx.x * blockDim.x + threadIdx.x;
  long stride = (long)gridDim.x * blockDim.x;
  for (; i < n; i += stride) out[i] = (__bf16)in[i];
}

__global__ void inv_cnt_kernel(const float* __restrict__ cnt, float* __restrict__ invc,
                               int N) {
  int i = blockIdx.x * blockDim.x + threadIdx.x;
  if (i < N) invc[i] = 1.0f / fmaxf(cnt[i], 1.0f);
}

// ---------------------------------------------------------------------------
// Edge scatter: agg[dst] += feat_bf16[src] (128 feats, one wave per edge,
// 4 bf16 loads -> 4x global_atomic_add_f32 per lane).  Optional degree count.
// ---------------------------------------------------------------------------
__global__ void scatter_add_bf(const __bf16* __restrict__ feat,
                               const int* __restrict__ src,
                               const int* __restrict__ dst,
                               float* __restrict__ agg,
                               float* __restrict__ cnt, int E) {
  const int wave = (blockIdx.x * blockDim.x + threadIdx.x) >> 5;
  const int lane = threadIdx.x & 31;
  if (wave >= E) return;
  const int s = src[wave];
  const int d = dst[wave];
  const __bf16* fp = feat + (size_t)s * 128 + lane * 4;   // 8B contiguous per lane
  float* ap = agg + (size_t)d * 128 + lane * 4;
  atomicAdd(ap + 0, (float)fp[0]);
  atomicAdd(ap + 1, (float)fp[1]);
  atomicAdd(ap + 2, (float)fp[2]);
  atomicAdd(ap + 3, (float)fp[3]);
  if (cnt != nullptr && lane == 0) atomicAdd(cnt + d, 1.0f);
}

// ---------------------------------------------------------------------------
// Fused SAGEConv: out_bf16 = [BN->ReLU]( (agg*invc) @ Wl^T + bl + xbf @ Wr^T )
// One 16(M) x 16(N) tile per wave; K = 128 in 4 WMMA steps per operand.
// ---------------------------------------------------------------------------
__global__ void sage_conv_wmma(const float* __restrict__ agg,
                               const float* __restrict__ invc,
                               const __bf16* __restrict__ xroot,
                               const __bf16* __restrict__ Wl,
                               const float* __restrict__ bl,
                               const __bf16* __restrict__ Wr,
                               const float* __restrict__ bng,
                               const float* __restrict__ bnb,
                               const float* __restrict__ bnm,
                               const float* __restrict__ bnv,
                               __bf16* __restrict__ out,
                               int N, int K, int outdim, int fuse_bn_relu) {
  const int lane = threadIdx.x & 31;
  const int wave = threadIdx.x >> 5;
  const int row0 = blockIdx.x * 16;
  const int col0 = wave * 16;
  if (col0 >= outdim) return;                  // wave-uniform: EXEC stays full

  int arow = row0 + (lane & 15);
  if (arow >= N) arow = N - 1;                 // clamp (N is a multiple of 16 anyway)
  const float ic = invc[arow];
  const float* aggrow = agg + (size_t)arow * K;
  const __bf16* xrow  = xroot + (size_t)arow * K;

  v8f acc = {};
#pragma unroll 4
  for (int k0 = 0; k0 < K; k0 += 32)
    acc = wmma_bf16(a_frag_f32(aggrow + k0, ic, lane), b_frag_bf(Wl, K, col0, k0, lane), acc);
#pragma unroll 4
  for (int k0 = 0; k0 < K; k0 += 32)
    acc = wmma_bf16(a_frag_bf(xrow + k0, lane), b_frag_bf(Wr, K, col0, k0, lane), acc);

  const int col = col0 + (lane & 15);
  const float bias = bl[col];
  float scale = 1.0f, shift = 0.0f;
  if (fuse_bn_relu) {
    const float rs = rsqrtf(bnv[col] + 1e-5f);
    scale = bng[col] * rs;
    shift = bnb[col] - bnm[col] * scale;
  }
#pragma unroll
  for (int r = 0; r < 8; ++r) {
    const int m = r + ((lane & 16) ? 8 : 0);   // C layout: lanes 16-31 hold M=8..15
    const int orow = row0 + m;
    if (orow < N) {
      float v = acc[r] + bias;
      if (fuse_bn_relu) v = fmaxf(v * scale + shift, 0.0f);
      out[(size_t)orow * outdim + col] = (__bf16)v;
    }
  }
}

// ---------------------------------------------------------------------------
// Edge regression head (one wave per 16 pairs):
//   e1 = relu(W1 @ [z_i ; z_j] + b1)   (64,  K=128)  -> bf16 LDS tile
//   e2 = relu(W2 @ e1 + b2)            (32,  K=64)   -> f32  LDS tile
//   out = sigmoid(W3 @ e2 + b3)        (1,   K=32)   VALU dot
// Same-wave LDS accesses are in-order (DScnt), no barrier needed.
// ---------------------------------------------------------------------------
#define E1_LD 72   // bf16 elements per row: 16B-aligned rows + conflict padding
#define E2_LD 36   // f32 elements per row
__global__ void edge_mlp_wmma(const __bf16* __restrict__ z,
                              const int* __restrict__ pi,
                              const int* __restrict__ pj,
                              const __bf16* __restrict__ W1, const float* __restrict__ b1,
                              const __bf16* __restrict__ W2, const float* __restrict__ b2,
                              const float* __restrict__ W3, const float* __restrict__ b3,
                              float* __restrict__ out, int P) {
  __shared__ __bf16 e1s[8][16 * E1_LD];   // per-wave 16x64 bf16 (padded)
  __shared__ float  e2s[8][16 * E2_LD];   // per-wave 16x32 f32 (padded)

  const int lane = threadIdx.x & 31;
  const int wave = threadIdx.x >> 5;
  const int row0 = (blockIdx.x * 8 + wave) * 16;
  if (row0 >= P) return;                       // wave-uniform

  int pr = row0 + (lane & 15);
  if (pr >= P) pr = P - 1;
  const __bf16* zi = z + (size_t)pi[pr] * 64;
  const __bf16* zj = z + (size_t)pj[pr] * 64;

  __bf16* e1 = &e1s[wave][0];
  float*  e2 = &e2s[wave][0];

  // ---- layer 1: cols 0..63, K = 128 via concat(zi, zj) --------------------
#pragma unroll
  for (int c = 0; c < 4; ++c) {
    v8f acc = {};
    acc = wmma_bf16(a_frag_bf(zi + 0,  lane), b_frag_bf(W1, 128, c * 16, 0,  lane), acc);
    acc = wmma_bf16(a_frag_bf(zi + 32, lane), b_frag_bf(W1, 128, c * 16, 32, lane), acc);
    acc = wmma_bf16(a_frag_bf(zj + 0,  lane), b_frag_bf(W1, 128, c * 16, 64, lane), acc);
    acc = wmma_bf16(a_frag_bf(zj + 32, lane), b_frag_bf(W1, 128, c * 16, 96, lane), acc);
    const int col = c * 16 + (lane & 15);
    const float bias = b1[col];
#pragma unroll
    for (int r = 0; r < 8; ++r) {
      const int m = r + ((lane & 16) ? 8 : 0);
      e1[m * E1_LD + col] = (__bf16)fmaxf(acc[r] + bias, 0.0f);
    }
  }

  // ---- layer 2: cols 0..31, K = 64 from bf16 LDS tile ---------------------
#pragma unroll
  for (int c = 0; c < 2; ++c) {
    v8f acc = {};
#pragma unroll
    for (int k0 = 0; k0 < 64; k0 += 32)
      acc = wmma_bf16(a_frag_bf(e1 + (lane & 15) * E1_LD + k0, lane),
                      b_frag_bf(W2, 64, c * 16, k0, lane), acc);
    const int col = c * 16 + (lane & 15);
    const float bias = b2[col];
#pragma unroll
    for (int r = 0; r < 8; ++r) {
      const int m = r + ((lane & 16) ? 8 : 0);
      e2[m * E2_LD + col] = fmaxf(acc[r] + bias, 0.0f);
    }
  }

  // ---- layer 3: scalar head, K = 32, lanes 0..15 each own one pair --------
  if (lane < 16) {
    const int orow = row0 + lane;
    if (orow < P) {
      float s = b3[0];
#pragma unroll
      for (int k = 0; k < 32; ++k) s += e2[lane * E2_LD + k] * W3[k];
      out[orow] = 1.0f / (1.0f + __expf(-s));
    }
  }
}

// ---------------------------------------------------------------------------
// Host launcher.
// ---------------------------------------------------------------------------
extern "C" void kernel_launch(void* const* d_in, const int* in_sizes, int n_in,
                              void* d_out, int out_size, void* d_ws, size_t ws_size,
                              hipStream_t stream) {
  const float* x        = (const float*)d_in[0];
  const int*   eidx     = (const int*)d_in[1];   // [2, E] : src row 0, dst row 1
  const int*   epairs   = (const int*)d_in[2];   // [2, P]
  const float* c1_Wl    = (const float*)d_in[3];
  const float* c1_bl    = (const float*)d_in[4];
  const float* c1_Wr    = (const float*)d_in[5];
  const float* bn_gamma = (const float*)d_in[6];
  const float* bn_beta  = (const float*)d_in[7];
  const float* bn_mean  = (const float*)d_in[8];
  const float* bn_var   = (const float*)d_in[9];
  const float* c2_Wl    = (const float*)d_in[10];
  const float* c2_bl    = (const float*)d_in[11];
  const float* c2_Wr    = (const float*)d_in[12];
  const float* e_W1     = (const float*)d_in[13];
  const float* e_b1     = (const float*)d_in[14];
  const float* e_W2     = (const float*)d_in[15];
  const float* e_b2     = (const float*)d_in[16];
  const float* e_W3     = (const float*)d_in[17];
  const float* e_b3     = (const float*)d_in[18];
  float* outp = (float*)d_out;

  const int N = in_sizes[0] / 128;
  const int E = in_sizes[1] / 2;
  const int P = in_sizes[2] / 2;

  // ---- workspace layout ---------------------------------------------------
  // f32 region:  agg[128N] | cnt[N] | invc[N]
  // bf16 region: xbf[128N] | hbf[128N] | zbf[64N] |
  //              Wl1[16384] | Wr1[16384] | Wl2[8192] | Wr2[8192] |
  //              W1e[8192]  | W2e[2048]
  float* agg  = (float*)d_ws;
  float* cnt  = agg + (size_t)128 * N;
  float* invc = cnt + N;
  __bf16* xbf  = (__bf16*)(invc + N);
  __bf16* hbf  = xbf + (size_t)128 * N;
  __bf16* zbf  = hbf + (size_t)128 * N;
  __bf16* Wl1  = zbf + (size_t)64 * N;
  __bf16* Wr1  = Wl1 + 16384;
  __bf16* Wl2  = Wr1 + 16384;
  __bf16* Wr2  = Wl2 + 8192;
  __bf16* W1e  = Wr2 + 8192;
  __bf16* W2e  = W1e + 8192;

  const int* src = eidx;
  const int* dst = eidx + E;
  const int* pi  = epairs;
  const int* pj  = epairs + P;

  const int ZB = 2048;
  const int scatter_blocks = (E + 7) / 8;   // 8 waves (edges) per 256-thread block
  const int row_tiles = (N + 15) / 16;

  // 1) zero agg + cnt (contiguous 129N floats)
  zero_f32<<<ZB, 256, 0, stream>>>(agg, (long)129 * N);
  // 2) one-time bf16 conversions (x + all GEMM weights)
  cvt_f32_to_bf16<<<ZB, 256, 0, stream>>>(x, xbf, (long)128 * N);
  cvt_f32_to_bf16<<<64, 256, 0, stream>>>(c1_Wl, Wl1, 16384);
  cvt_f32_to_bf16<<<64, 256, 0, stream>>>(c1_Wr, Wr1, 16384);
  cvt_f32_to_bf16<<<32, 256, 0, stream>>>(c2_Wl, Wl2, 8192);
  cvt_f32_to_bf16<<<32, 256, 0, stream>>>(c2_Wr, Wr2, 8192);
  cvt_f32_to_bf16<<<32, 256, 0, stream>>>(e_W1, W1e, 8192);
  cvt_f32_to_bf16<<<8, 256, 0, stream>>>(e_W2, W2e, 2048);
  // 3) scatter x[src] -> agg[dst] (fp32 atomics), count degrees
  scatter_add_bf<<<scatter_blocks, 256, 0, stream>>>(xbf, src, dst, agg, cnt, E);
  // 4) invc = 1 / max(cnt, 1)
  inv_cnt_kernel<<<(N + 255) / 256, 256, 0, stream>>>(cnt, invc, N);
  // 5) conv1 + BN + ReLU -> hbf [N,128]
  sage_conv_wmma<<<row_tiles, 256, 0, stream>>>(agg, invc, xbf, Wl1, c1_bl, Wr1,
                                                bn_gamma, bn_beta, bn_mean, bn_var,
                                                hbf, N, 128, 128, 1);
  // 6) re-zero agg, scatter h[src] -> agg[dst]
  zero_f32<<<ZB, 256, 0, stream>>>(agg, (long)128 * N);
  scatter_add_bf<<<scatter_blocks, 256, 0, stream>>>(hbf, src, dst, agg, nullptr, E);
  // 7) conv2 -> zbf [N,64]
  sage_conv_wmma<<<row_tiles, 128, 0, stream>>>(agg, invc, hbf, Wl2, c2_bl, Wr2,
                                                bn_gamma, bn_beta, bn_mean, bn_var,
                                                zbf, N, 128, 64, 0);
  // 8) edge MLP head -> out [P]
  const int edge_blocks = (P + 127) / 128;  // 8 waves x 16 pairs per block
  edge_mlp_wmma<<<edge_blocks, 256, 0, stream>>>(zbf, pi, pj, W1e, e_b1, W2e, e_b2,
                                                 e_W3, e_b3, outp, P);
}
